// BiLSTMForTokenClassificationWithCRF_57586921504773
// MI455X (gfx1250) — matrix operand order repaired and checked by reading, not compiled, not run on hardware
//
#include <hip/hip_runtime.h>
#include <hip/hip_bf16.h>
#include <math.h>

// ---------------------------------------------------------------------------
// Problem constants
// ---------------------------------------------------------------------------
#define BB   64
#define SS   512
#define II   300
#define KI   320          // I padded to multiple of 32
#define HH   512
#define G4H  2048         // 4*H
#define NIN  4096         // both directions' gates
#define H2   1024         // 2*H
#define TT   25
#define TPAD 32

typedef __bf16 bhalf;
typedef bhalf v16bf __attribute__((ext_vector_type(16)));
typedef bhalf v8bf  __attribute__((ext_vector_type(8)));
typedef float v8f   __attribute__((ext_vector_type(8)));

// ---------------------------------------------------------------------------
// Workspace layout (bytes, 256-aligned regions)
// ---------------------------------------------------------------------------
#define OFF_XBF   ((size_t)0)                         // 32768*320 bf16
#define OFF_WIH   ((size_t)20971520)                  // 4096*320 bf16 (f|b gates)
#define OFF_WHH   ((size_t)23592960)                  // 2*2048*512 bf16
#define OFF_WC    ((size_t)27787264)                  // 32*1024 bf16 (T padded)
#define OFF_LEN   ((size_t)27852800)                  // 64 int
#define OFF_ACC   ((size_t)27853056)                  // 2 float accumulators
#define OFF_H32   ((size_t)27853312)                  // 2*64*512 f32
#define OFF_C32   ((size_t)28115456)                  // 2*64*512 f32
#define OFF_HBF   ((size_t)28377600)                  // 2(parity)*2*64*512 bf16
#define OFF_SEQ   ((size_t)28639744)                  // 32768*1024 bf16
#define OFF_HIST  ((size_t)95748608)                  // 64*512*32 u8 viterbi hist
#define OFF_PRE   ((size_t)96797184)                  // 2*512*64*2048 f32
#define HBUF      ((size_t)(2*BB*HH))                 // bf16 elems per parity

// ---------------------------------------------------------------------------
// WMMA helpers (CDNA5 wave32, 16x16x32 bf16 -> f32)
// ---------------------------------------------------------------------------
__device__ __forceinline__ v8f vz8() {
  v8f z;
#pragma unroll
  for (int i = 0; i < 8; ++i) z[i] = 0.f;
  return z;
}

__device__ __forceinline__ v8f wmma_bf16(v16bf a, v16bf b, v8f c) {
  return __builtin_amdgcn_wmma_f32_16x16x32_bf16(false, a, false, b, (short)0, c,
                                                 false, false);
}

// A fragment: 16x32 (MxK) bf16, row-major source [rows][ld].
// lanes 0-15 hold K 0..15; lanes 16-31 hold K 16..31.
__device__ __forceinline__ v16bf load_frag_a(const bhalf* base, int ld, int m0, int k0) {
  const int lane = threadIdx.x & 31;
  const int row  = m0 + (lane & 15);
  const int kh   = lane >> 4;                   // 0 or 1
  const bhalf* p = base + (size_t)row * ld + k0 + kh * 16;
  return *(const v16bf*)p;
}

// B fragment: 32x16 (KxN) bf16 where B[k][n] = W[n0+n][k0+k], W row-major [N][ld].
// lanes 0-15: K 0..15 (col n=lane); lanes 16-31: K 16..31 (col n=lane-16).
__device__ __forceinline__ v16bf load_frag_b(const bhalf* base, int ld, int n0, int k0) {
  const int lane = threadIdx.x & 31;
  const int col  = n0 + (lane & 15);
  const int kh   = lane >> 4;
  const bhalf* p = base + (size_t)col * ld + k0 + kh * 16;
  return *(const v16bf*)p;
}

// ---------------------------------------------------------------------------
// Kernel 0: precision conversion / padding / state init / lengths
// ---------------------------------------------------------------------------
__global__ void prep_kernel(const float* __restrict__ x,
                            const float* __restrict__ wihf, const float* __restrict__ wihb,
                            const float* __restrict__ whhf, const float* __restrict__ whhb,
                            const float* __restrict__ wcf,  const int* __restrict__ am,
                            bhalf* xbf, bhalf* wih, bhalf* whh, bhalf* wc,
                            int* lenbuf, float* accum,
                            float* h32, float* c32, bhalf* hbf) {
  const size_t idx = (size_t)blockIdx.x * blockDim.x + threadIdx.x;
  if (idx < (size_t)BB * SS * KI) {                       // x -> bf16, K pad
    int r = (int)(idx / KI), c = (int)(idx % KI);
    xbf[idx] = (c < II) ? (bhalf)x[(size_t)r * II + c] : (bhalf)0.f;
  }
  if (idx < (size_t)NIN * KI) {                           // [Wihf;Wihb] -> bf16
    int r = (int)(idx / KI), c = (int)(idx % KI);
    float v = 0.f;
    if (c < II) v = (r < G4H) ? wihf[(size_t)r * II + c]
                              : wihb[(size_t)(r - G4H) * II + c];
    wih[idx] = (bhalf)v;
  }
  if (idx < (size_t)2 * G4H * HH) {                       // W_hh both dirs
    const size_t half = (size_t)G4H * HH;
    whh[idx] = (bhalf)((idx < half) ? whhf[idx] : whhb[idx - half]);
  }
  if (idx < (size_t)TPAD * H2) {                          // Wc, rows >= 25 zero
    int r = (int)(idx / H2), c = (int)(idx % H2);
    wc[idx] = (r < TT) ? (bhalf)wcf[(size_t)r * H2 + c] : (bhalf)0.f;
  }
  if (idx < (size_t)2 * BB * HH) {                        // h/c state, both hbf parities
    h32[idx] = 0.f; c32[idx] = 0.f;
    hbf[idx] = (bhalf)0.f; hbf[idx + 2 * BB * HH] = (bhalf)0.f;
  }
  if (idx < BB) {                                         // sequence lengths
    int s = 0;
    for (int t = 0; t < SS; ++t) s += (am[idx * SS + t] > 0) ? 1 : 0;
    lenbuf[idx] = s;
  }
  if (idx < 2) accum[idx] = 0.f;
}

// ---------------------------------------------------------------------------
// Kernel 1: pre[d][s][b][g] = x @ [Wihf;Wihb]^T    (M=32768, N=4096, K=320)
// block = 8 waves; wave tile 16x64; block tile 128x64.
// Ping-pong double-buffered k-loop: loads write directly into the fragment
// set consumed two WMMA-groups later -> no register rotation moves.
// ---------------------------------------------------------------------------
__global__ __launch_bounds__(256)
void gemm_input_kernel(const bhalf* __restrict__ xbf, const bhalf* __restrict__ wih,
                       float* __restrict__ pre) {
  const int wave = threadIdx.x >> 5;
  const int lane = threadIdx.x & 31;
  const int m0 = blockIdx.y * 128 + wave * 16;
  const int n0 = blockIdx.x * 64;
  v8f acc[4];
#pragma unroll
  for (int j = 0; j < 4; ++j) acc[j] = vz8();

  v16bf a0 = load_frag_a(xbf, KI, m0, 0);
  v16bf a1 = load_frag_a(xbf, KI, m0, 32);
  v16bf b0[4], b1[4];
#pragma unroll
  for (int j = 0; j < 4; ++j) b0[j] = load_frag_b(wih, KI, n0 + j * 16, 0);
#pragma unroll
  for (int j = 0; j < 4; ++j) b1[j] = load_frag_b(wih, KI, n0 + j * 16, 32);

  for (int k0 = 0; k0 < KI - 64; k0 += 64) {
#pragma unroll
    for (int j = 0; j < 4; ++j) acc[j] = wmma_bf16(a0, b0[j], acc[j]);
    a0 = load_frag_a(xbf, KI, m0, k0 + 64);
#pragma unroll
    for (int j = 0; j < 4; ++j) b0[j] = load_frag_b(wih, KI, n0 + j * 16, k0 + 64);
#pragma unroll
    for (int j = 0; j < 4; ++j) acc[j] = wmma_bf16(a1, b1[j], acc[j]);
    a1 = load_frag_a(xbf, KI, m0, k0 + 96);
#pragma unroll
    for (int j = 0; j < 4; ++j) b1[j] = load_frag_b(wih, KI, n0 + j * 16, k0 + 96);
  }
#pragma unroll
  for (int j = 0; j < 4; ++j) acc[j] = wmma_bf16(a0, b0[j], acc[j]);
#pragma unroll
  for (int j = 0; j < 4; ++j) acc[j] = wmma_bf16(a1, b1[j], acc[j]);

  const int mloc = (lane >> 4) * 8;
  const int n    = lane & 15;
#pragma unroll
  for (int j = 0; j < 4; ++j) {
#pragma unroll
    for (int i = 0; i < 8; ++i) {
      const int r = m0 + mloc + i;            // row in x ordering: b*512+s
      const int g = n0 + j * 16 + n;          // 0..4095
      const int b_ = r >> 9, s = r & (SS - 1);
      const int d = g >> 11, gg = g & (G4H - 1);
      pre[(((size_t)d * SS + s) * BB + b_) * G4H + gg] = acc[j][i];
    }
  }
}

// ---------------------------------------------------------------------------
// Kernel 2 (x512): one LSTM timestep, both directions.
// grid (16, 1, 2): 16 column tiles of 32 h-units, z = direction.
// gates = pre[t] + h @ W_hh^T + bias; fused cell + masked update + scatter.
// Ping-pong pipelined k-loop (serial critical path).
// ---------------------------------------------------------------------------
__global__ __launch_bounds__(256)
void lstm_step_kernel(int t,
                      const float* __restrict__ pre,
                      const float* __restrict__ bias_f, const float* __restrict__ bias_b,
                      const bhalf* __restrict__ whh,
                      const int* __restrict__ am, const int* __restrict__ lenbuf,
                      float* __restrict__ h32, float* __restrict__ c32,
                      const bhalf* __restrict__ hin, bhalf* __restrict__ hout,
                      bhalf* __restrict__ seq) {
  const int d     = blockIdx.z;
  const int jcol0 = blockIdx.x * 32;
  const int wave  = threadIdx.x >> 5;
  const int lane  = threadIdx.x & 31;
  const int mt    = wave >> 1;                  // batch tile 0..3
  const int nt    = wave & 1;                   // 16-col sub-tile 0..1

  __shared__ float gl[4][BB][32];               // i/f/g/o gate tiles, 32 KB

  const bhalf* hA   = hin + (size_t)d * BB * HH;
  const bhalf* wB   = whh + (size_t)d * G4H * HH;
  const float* bias = (d == 0) ? bias_f : bias_b;
  const int nb0 = jcol0 + nt * 16;              // base col within each gate block

  v8f acc[4];
#pragma unroll
  for (int g = 0; g < 4; ++g) acc[g] = vz8();

  v16bf a0 = load_frag_a(hA, HH, mt * 16, 0);
  v16bf a1 = load_frag_a(hA, HH, mt * 16, 32);
  v16bf b0[4], b1[4];
#pragma unroll
  for (int g = 0; g < 4; ++g) b0[g] = load_frag_b(wB, HH, g * HH + nb0, 0);
#pragma unroll
  for (int g = 0; g < 4; ++g) b1[g] = load_frag_b(wB, HH, g * HH + nb0, 32);

  for (int k0 = 0; k0 < HH - 64; k0 += 64) {
#pragma unroll
    for (int g = 0; g < 4; ++g) acc[g] = wmma_bf16(a0, b0[g], acc[g]);
    a0 = load_frag_a(hA, HH, mt * 16, k0 + 64);
#pragma unroll
    for (int g = 0; g < 4; ++g) b0[g] = load_frag_b(wB, HH, g * HH + nb0, k0 + 64);
#pragma unroll
    for (int g = 0; g < 4; ++g) acc[g] = wmma_bf16(a1, b1[g], acc[g]);
    a1 = load_frag_a(hA, HH, mt * 16, k0 + 96);
#pragma unroll
    for (int g = 0; g < 4; ++g) b1[g] = load_frag_b(wB, HH, g * HH + nb0, k0 + 96);
  }
#pragma unroll
  for (int g = 0; g < 4; ++g) acc[g] = wmma_bf16(a0, b0[g], acc[g]);
#pragma unroll
  for (int g = 0; g < 4; ++g) acc[g] = wmma_bf16(a1, b1[g], acc[g]);

  const int mloc = (lane >> 4) * 8;
  const int n    = lane & 15;
#pragma unroll
  for (int g = 0; g < 4; ++g) {
#pragma unroll
    for (int i = 0; i < 8; ++i) {
      const int bi  = mt * 16 + mloc + i;
      const int jj  = nt * 16 + n;
      const int gg  = g * HH + jcol0 + jj;
      const int len = lenbuf[bi];
      const int row = (d == 0) ? t : ((t < len) ? (len - 1 - t) : t);
      const float p = pre[(((size_t)d * SS + row) * BB + bi) * G4H + gg];
      gl[g][bi][jj] = acc[g][i] + p + bias[gg];
    }
  }
  __syncthreads();

  for (int e = threadIdx.x; e < BB * 32; e += 256) {
    const int bi = e >> 5;
    const int jj = e & 31;
    const int j  = jcol0 + jj;
    const float gi = gl[0][bi][jj];
    const float gf = gl[1][bi][jj];
    const float gc = gl[2][bi][jj];
    const float go = gl[3][bi][jj];
    const size_t hoff = (size_t)d * BB * HH + (size_t)bi * HH + j;
    const float c_old = c32[hoff];
    const float h_old = h32[hoff];
    const float si = 1.f / (1.f + expf(-gi));
    const float sf = 1.f / (1.f + expf(-gf));
    const float so = 1.f / (1.f + expf(-go));
    const float c_new = sf * c_old + si * tanhf(gc);
    const float h_new = so * tanhf(c_new);
    const int   m  = am[bi * SS + t] > 0;
    const float hk = m ? h_new : h_old;
    h32[hoff] = hk;
    c32[hoff] = m ? c_new : c_old;
    hout[hoff] = (bhalf)hk;
    const int len = lenbuf[bi];
    const int orow = (d == 0) ? t : ((t < len) ? (len - 1 - t) : t);
    seq[((size_t)bi * SS + orow) * H2 + d * HH + j] = (bhalf)(m ? h_new : 0.f);
  }
}

// ---------------------------------------------------------------------------
// Kernel 3: logits = seq @ Wc^T + bc   (M=32768, N=32(pad 25), K=1024)
// ---------------------------------------------------------------------------
__global__ __launch_bounds__(256)
void classifier_kernel(const bhalf* __restrict__ seq, const bhalf* __restrict__ wc,
                       const float* __restrict__ bc, float* __restrict__ logits) {
  const int wave = threadIdx.x >> 5;
  const int lane = threadIdx.x & 31;
  const int m0 = blockIdx.x * 128 + wave * 16;
  v8f acc[2];
  acc[0] = vz8(); acc[1] = vz8();

  v16bf a0 = load_frag_a(seq, H2, m0, 0);
  v16bf a1 = load_frag_a(seq, H2, m0, 32);
  v16bf b0[2], b1[2];
#pragma unroll
  for (int j = 0; j < 2; ++j) b0[j] = load_frag_b(wc, H2, j * 16, 0);
#pragma unroll
  for (int j = 0; j < 2; ++j) b1[j] = load_frag_b(wc, H2, j * 16, 32);

  for (int k0 = 0; k0 < H2 - 64; k0 += 64) {
#pragma unroll
    for (int j = 0; j < 2; ++j) acc[j] = wmma_bf16(a0, b0[j], acc[j]);
    a0 = load_frag_a(seq, H2, m0, k0 + 64);
#pragma unroll
    for (int j = 0; j < 2; ++j) b0[j] = load_frag_b(wc, H2, j * 16, k0 + 64);
#pragma unroll
    for (int j = 0; j < 2; ++j) acc[j] = wmma_bf16(a1, b1[j], acc[j]);
    a1 = load_frag_a(seq, H2, m0, k0 + 96);
#pragma unroll
    for (int j = 0; j < 2; ++j) b1[j] = load_frag_b(wc, H2, j * 16, k0 + 96);
  }
#pragma unroll
  for (int j = 0; j < 2; ++j) acc[j] = wmma_bf16(a0, b0[j], acc[j]);
#pragma unroll
  for (int j = 0; j < 2; ++j) acc[j] = wmma_bf16(a1, b1[j], acc[j]);

  const int mloc = (lane >> 4) * 8;
  const int n    = lane & 15;
#pragma unroll
  for (int j = 0; j < 2; ++j) {
    const int tgt = j * 16 + n;
    if (tgt < TT) {
#pragma unroll
      for (int i = 0; i < 8; ++i) {
        const int r = m0 + mloc + i;
        logits[(size_t)r * TT + tgt] = acc[j][i] + bc[tgt];
      }
    }
  }
}

// ---------------------------------------------------------------------------
// Kernel 4: CRF negative log-likelihood (token mean numerator/denominator).
// One wave per batch; mask is a contiguous prefix -> early exit at len.
// ---------------------------------------------------------------------------
__global__ void crf_nll_kernel(const float* __restrict__ lo, const int* __restrict__ am,
                               const int* __restrict__ labels, const int* __restrict__ lenbuf,
                               const float* __restrict__ start, const float* __restrict__ end,
                               const float* __restrict__ trans, float* accum) {
  const int b = blockIdx.x;
  const int lane = threadIdx.x;
  __shared__ float sc[TPAD];
  __shared__ float tr[TT * TT];
  __shared__ float red[32];
  for (int i = lane; i < TT * TT; i += 32) tr[i] = trans[i];
  const int len = lenbuf[b];
  const float* em = lo + (size_t)b * SS * TT;
  const int* tg = labels + b * SS;
  __syncthreads();

  float part = 0.f;                                // numerator (strided over t)
  for (int t = lane; t < len; t += 32) {
    const int ct = tg[t];
    if (t == 0) part += start[ct] + em[ct];
    else        part += tr[tg[t - 1] * TT + ct] + em[t * TT + ct];
  }
  red[lane] = part;

  if (lane < TT) sc[lane] = start[lane] + em[lane];
  __syncthreads();
  for (int t = 1; t < len; ++t) {                  // forward logsumexp recursion
    float nv = 0.f;
    if (lane < TT) {
      float mx = -3.4e38f;
      for (int i = 0; i < TT; ++i) mx = fmaxf(mx, sc[i] + tr[i * TT + lane]);
      float s = 0.f;
      for (int i = 0; i < TT; ++i) s += expf(sc[i] + tr[i * TT + lane] - mx);
      nv = mx + logf(s) + em[t * TT + lane];
    }
    __syncthreads();
    if (lane < TT) sc[lane] = nv;
    __syncthreads();
  }
  if (lane == 0) {
    float num = 0.f;
    for (int i = 0; i < 32; ++i) num += red[i];
    num += end[tg[len - 1]];
    float mx = -3.4e38f;
    for (int j = 0; j < TT; ++j) mx = fmaxf(mx, sc[j] + end[j]);
    float s = 0.f;
    for (int j = 0; j < TT; ++j) s += expf(sc[j] + end[j] - mx);
    const float denom = mx + logf(s);
    atomicAdd(&accum[0], num - denom);
    atomicAdd(&accum[1], (float)len);
  }
}

// ---------------------------------------------------------------------------
// Kernel 5: CRF Viterbi decode + backtrack + pred_mask output.
// ---------------------------------------------------------------------------
__global__ void crf_decode_kernel(const float* __restrict__ lo, const int* __restrict__ am,
                                  const int* __restrict__ lenbuf,
                                  const float* __restrict__ start, const float* __restrict__ end,
                                  const float* __restrict__ trans,
                                  unsigned char* __restrict__ hist,
                                  float* __restrict__ tags_out, float* __restrict__ mask_out) {
  const int b = blockIdx.x;
  const int lane = threadIdx.x;
  __shared__ float sc[TPAD];
  __shared__ float tr[TT * TT];
  __shared__ int lastTag;
  for (int i = lane; i < TT * TT; i += 32) tr[i] = trans[i];
  const int len = lenbuf[b];
  const float* em = lo + (size_t)b * SS * TT;
  if (lane < TT) sc[lane] = start[lane] + em[lane];
  __syncthreads();
  for (int t = 1; t < len; ++t) {
    float best = -3.4e38f; int bp = 0;
    if (lane < TT) {
      for (int i = 0; i < TT; ++i) {
        const float v = sc[i] + tr[i * TT + lane];
        if (v > best) { best = v; bp = i; }
      }
      best += em[t * TT + lane];
    }
    __syncthreads();
    if (lane < TT) {
      sc[lane] = best;
      hist[((size_t)b * SS + (t - 1)) * TPAD + lane] = (unsigned char)bp;
    }
    __syncthreads();
  }
  if (lane == 0) {
    float best = -3.4e38f; int bt = 0;
    for (int j = 0; j < TT; ++j) {
      const float v = sc[j] + end[j];
      if (v > best) { best = v; bt = j; }
    }
    lastTag = bt;
  }
  __syncthreads();
  for (int t = lane; t < SS; t += 32)
    mask_out[b * SS + t] = (am[b * SS + t] > 0) ? 1.f : 0.f;
  if (lane == 0) {
    int tag = lastTag;
    tags_out[b * SS + (SS - 1)] = (float)tag;
    for (int k = SS - 2; k >= 0; --k) {
      if (am[b * SS + k + 1] > 0)
        tag = hist[((size_t)b * SS + k) * TPAD + tag];
      tags_out[b * SS + k] = (float)tag;
    }
  }
}

__global__ void finalize_kernel(const float* accum, float* out) {
  if (blockIdx.x == 0 && threadIdx.x == 0) out[0] = -(accum[0] / accum[1]);
}

// ---------------------------------------------------------------------------
// Host launcher (all launches on `stream`; graph-capture safe)
// ---------------------------------------------------------------------------
extern "C" void kernel_launch(void* const* d_in, const int* in_sizes, int n_in,
                              void* d_out, int out_size, void* d_ws, size_t ws_size,
                              hipStream_t stream) {
  (void)in_sizes; (void)n_in; (void)out_size; (void)ws_size;
  const float* x    = (const float*)d_in[0];
  const float* Wihf = (const float*)d_in[1];
  const float* Whhf = (const float*)d_in[2];
  const float* bf_  = (const float*)d_in[3];
  const float* Wihb = (const float*)d_in[4];
  const float* Whhb = (const float*)d_in[5];
  const float* bb_  = (const float*)d_in[6];
  const float* Wcm  = (const float*)d_in[7];
  const float* bc   = (const float*)d_in[8];
  const float* cs   = (const float*)d_in[9];
  const float* ce   = (const float*)d_in[10];
  const float* ct   = (const float*)d_in[11];
  const int*   am   = (const int*)d_in[12];
  const int*   lbl  = (const int*)d_in[13];

  float* out = (float*)d_out;
  char*  ws  = (char*)d_ws;
  bhalf* xbf   = (bhalf*)(ws + OFF_XBF);
  bhalf* wih   = (bhalf*)(ws + OFF_WIH);
  bhalf* whh   = (bhalf*)(ws + OFF_WHH);
  bhalf* wc    = (bhalf*)(ws + OFF_WC);
  int*   len   = (int*)  (ws + OFF_LEN);
  float* accum = (float*)(ws + OFF_ACC);
  float* h32   = (float*)(ws + OFF_H32);
  float* c32   = (float*)(ws + OFF_C32);
  bhalf* hbf   = (bhalf*)(ws + OFF_HBF);
  bhalf* seq   = (bhalf*)(ws + OFF_SEQ);
  unsigned char* hist = (unsigned char*)(ws + OFF_HIST);
  float* pre   = (float*)(ws + OFF_PRE);

  float* logits   = out + 1;
  float* tags_out = out + 1 + (size_t)BB * SS * TT;
  float* mask_out = tags_out + (size_t)BB * SS;

  // 0) convert to bf16 / pad / zero state / lengths
  prep_kernel<<<(BB * SS * KI + 255) / 256, 256, 0, stream>>>(
      x, Wihf, Wihb, Whhf, Whhb, Wcm, am,
      xbf, wih, whh, wc, len, accum, h32, c32, hbf);

  // 1) input projection for both directions (WMMA bf16)
  gemm_input_kernel<<<dim3(NIN / 64, (BB * SS) / 128), 256, 0, stream>>>(xbf, wih, pre);

  // 2) recurrence: 512 step kernels, h double-buffered in bf16
  for (int t = 0; t < SS; ++t) {
    lstm_step_kernel<<<dim3(HH / 32, 1, 2), 256, 0, stream>>>(
        t, pre, bf_, bb_, whh, am, len, h32, c32,
        hbf + (size_t)(t & 1) * HBUF, hbf + (size_t)((t + 1) & 1) * HBUF, seq);
  }

  // 3) classifier (WMMA bf16) -> logits straight into d_out
  classifier_kernel<<<(BB * SS) / 128, 256, 0, stream>>>(seq, wc, bc, logits);

  // 4) CRF loss + Viterbi decode + finalize
  crf_nll_kernel<<<BB, 32, 0, stream>>>(logits, am, lbl, len, cs, ce, ct, accum);
  crf_decode_kernel<<<BB, 32, 0, stream>>>(logits, am, len, cs, ce, ct, hist,
                                           tags_out, mask_out);
  finalize_kernel<<<1, 1, 0, stream>>>(accum, out);
}
